// SMoEAdapter_DOWN_33414845563681
// MI455X (gfx1250) — matrix operand order repaired and checked by reading, not compiled, not run on hardware
//
#include <hip/hip_runtime.h>
#include <math.h>

// ---------------- problem constants ----------------
#define HIDDEN 2048
#define DOWN   512
#define NEXP   16
#define TOPK   2
#define NTOK   8192            // B*S = 4*2048
#define PADB   144             // padded LDS row stride (bytes) for 128B rows

typedef __attribute__((ext_vector_type(16))) __bf16 v16bf;
typedef __attribute__((ext_vector_type(8)))  __bf16 v8bf;
typedef __attribute__((ext_vector_type(4)))  __bf16 v4bf;
typedef __attribute__((ext_vector_type(8)))  float  v8f;

// ---------------- CDNA5 async copy helpers (global -> LDS) ----------------
__device__ __forceinline__ void async_load_b128(unsigned int lds_off, const void* gptr) {
    // per-lane 16B copy: LDS[lds_off] = MEM[gptr]; tracked by ASYNCcnt
    asm volatile("global_load_async_to_lds_b128 %0, %1, off"
                 :: "v"(lds_off), "v"(gptr)
                 : "memory");
}
__device__ __forceinline__ void wait_async0() {
    asm volatile("s_wait_asynccnt 0x0" ::: "memory");
}

// stage a 64-row x 64-K bf16 chunk (8KB) into LDS at ldsBase, 4 x 16B per thread
template<int STRIDE>
__device__ __forceinline__ void stage_b_chunk(const __bf16* base, int k0,
                                              unsigned int ldsBase, int tid) {
    #pragma unroll
    for (int s = 0; s < 4; ++s) {
        const int linear = tid + s * 128;
        const int r      = linear >> 3;
        const int koff   = linear & 7;
        async_load_b128(ldsBase + (unsigned)(r * PADB + koff * 16),
                        base + (size_t)r * STRIDE + k0 + koff * 8);
    }
}

// ---------------- WMMA helpers ----------------
__device__ __forceinline__ v8f wmma_bf16(v16bf a, v16bf b, v8f c) {
    return __builtin_amdgcn_wmma_f32_16x16x32_bf16(false, a, false, b, (short)0, c,
                                                   false, false);
}

__device__ __forceinline__ v16bf make_afrag(const __bf16* p) {
    v8bf lo = *(const v8bf*)(p);
    v8bf hi = *(const v8bf*)(p + 8);
    return __builtin_shufflevector(lo, hi, 0, 1, 2, 3, 4, 5, 6, 7,
                                           8, 9, 10, 11, 12, 13, 14, 15);
}

// ---------------- f32 -> bf16 convert (vectorized) ----------------
__global__ __launch_bounds__(256) void cvt_bf16_kernel(const float* __restrict__ src,
                                                       __bf16* __restrict__ dst, int n4) {
    int i = blockIdx.x * 256 + threadIdx.x;
    if (i >= n4) return;
    const float4 f = *(const float4*)(src + (size_t)i * 4);
    v4bf o;
    o[0] = (__bf16)f.x; o[1] = (__bf16)f.y; o[2] = (__bf16)f.z; o[3] = (__bf16)f.w;
    *(v4bf*)(dst + (size_t)i * 4) = o;
}

// ---------------- init: zero expert counters ----------------
__global__ void init_kernel(int* counts) {
    if (threadIdx.x < NEXP) counts[threadIdx.x] = 0;
}

// ---------------- gate: logits, top-2 softmax, binning ----------------
// one block (16 waves) per token; wave w computes logit for expert w
__global__ __launch_bounds__(512) void gate_kernel(const float* __restrict__ x,
                                                   const float* __restrict__ gate_w,
                                                   int* __restrict__ counts,
                                                   int* __restrict__ tokenIds,
                                                   int* __restrict__ slotIds,
                                                   float* __restrict__ tokenProbs,
                                                   int* __restrict__ ti,
                                                   float* __restrict__ tp) {
    __shared__ float slog[NEXP];
    const int n    = blockIdx.x;
    const int wave = threadIdx.x >> 5;
    const int lane = threadIdx.x & 31;
    const float* xr = x + (size_t)n * HIDDEN;
    const float* gr = gate_w + (size_t)wave * HIDDEN;
    float s = 0.f;
    #pragma unroll 8
    for (int i = 0; i < HIDDEN / 32; ++i)
        s += xr[lane + 32 * i] * gr[lane + 32 * i];
    #pragma unroll
    for (int off = 16; off > 0; off >>= 1)
        s += __shfl_xor(s, off, 32);
    if (lane == 0) slog[wave] = s;
    __syncthreads();
    if (threadIdx.x == 0) {
        float v0 = -1e30f; int i0 = 0;
        #pragma unroll
        for (int e = 0; e < NEXP; ++e) if (slog[e] > v0) { v0 = slog[e]; i0 = e; }
        float v1 = -1e30f; int i1 = 0;
        #pragma unroll
        for (int e = 0; e < NEXP; ++e) if (e != i0 && slog[e] > v1) { v1 = slog[e]; i1 = e; }
        const float e1 = __expf(v1 - v0);
        const float p0 = 1.f / (1.f + e1);
        const float p1 = e1 * p0;
        const int   ie[TOPK] = { i0, i1 };
        const float pe[TOPK] = { p0, p1 };
        #pragma unroll
        for (int k = 0; k < TOPK; ++k) {
            const int e = ie[k];
            const int pos = atomicAdd(&counts[e], 1);
            const int idx = e * NTOK + pos;
            tokenIds[idx]   = n;
            slotIds[idx]    = n * TOPK + k;
            tokenProbs[idx] = pe[k];
            ti[n * TOPK + k] = e;
            tp[n * TOPK + k] = pe[k];
        }
    }
}

// ---------------- expert down-projection (gathered GEMM, double-buffered
// async-LDS-staged B) -------------------------------------------------------
// grid (E, DOWN/64, 128), block 128 (4 waves). Block covers a 64-wide N group
// of one expert; wave w handles m-tile (blockIdx.z*4 + w). B chunks (64 cols x
// 64 K, 8KB) ping-pong through LDS; chunk c+1 is DMA'd while chunk c computes.
__global__ __launch_bounds__(128) void down_gemm_kernel(const __bf16* __restrict__ xbf,
                                                        const __bf16* __restrict__ dwbf,
                                                        const float* __restrict__ down_b,
                                                        const int* __restrict__ counts,
                                                        const int* __restrict__ tokenIds,
                                                        const int* __restrict__ slotIds,
                                                        const float* __restrict__ tokenProbs,
                                                        float* __restrict__ hbuf) {
    __shared__ __align__(16) char sB[2][64 * PADB];
    const int e    = blockIdx.x;
    const int n0g  = blockIdx.y * 64;
    const int wv   = threadIdx.x >> 5;
    const int lane = threadIdx.x & 31;
    const int lm   = lane & 15;
    const int lh   = lane >> 4;
    const int count = counts[e];
    if (blockIdx.z * 4 * 16 >= count) return;         // uniform early exit
    const int mt     = blockIdx.z * 4 + wv;
    const bool active = (mt * 16) < count;
    const int binBase = e * NTOK;

    const int m   = mt * 16 + lm;
    const int tid = (active && m < count) ? tokenIds[binBase + m] : 0;
    const __bf16* arow  = xbf + (size_t)tid * HIDDEN + lh * 16;
    const __bf16* wbase = dwbf + ((size_t)e * DOWN + n0g) * HIDDEN;
    const unsigned int ldsBase0 = (unsigned int)(uintptr_t)(void*)sB[0];
    const unsigned int ldsBase1 = (unsigned int)(uintptr_t)(void*)sB[1];

    v8f acc[4] = {};
    stage_b_chunk<HIDDEN>(wbase, 0, ldsBase0, threadIdx.x);   // prologue: chunk 0
    for (int c = 0, k0 = 0; k0 < HIDDEN; k0 += 64, ++c) {
        wait_async0();           // own pieces of chunk c have landed
        __syncthreads();         // everyone's pieces visible; prior reads of the
                                 // other buffer are done -> safe to overwrite it
        if (k0 + 64 < HIDDEN)
            stage_b_chunk<HIDDEN>(wbase, k0 + 64,
                                  (c & 1) ? ldsBase0 : ldsBase1, threadIdx.x);
        const char* bb = sB[c & 1];
        if (k0 + 64 < HIDDEN) __builtin_prefetch(arow + k0 + 64, 0, 3);
        #pragma unroll
        for (int kk = 0; kk < 64; kk += 32) {
            const v16bf a = make_afrag(arow + k0 + kk);
            #pragma unroll
            for (int j = 0; j < 4; ++j) {
                const v16bf b = *(const v16bf*)(bb + (j * 16 + lm) * PADB + kk * 2 + lh * 32);
                acc[j] = wmma_bf16(a, b, acc[j]);
            }
        }
    }

    if (!active) return;
    // per-row routing metadata (shared across the 4 N sub-tiles)
    int   slot[8];
    float prob[8];
    #pragma unroll
    for (int r = 0; r < 8; ++r) {
        const int mm = mt * 16 + r + 8 * lh;
        const bool ok = mm < count;
        slot[r] = ok ? slotIds[binBase + mm] : -1;
        prob[r] = ok ? tokenProbs[binBase + mm] : 0.f;
    }
    #pragma unroll
    for (int j = 0; j < 4; ++j) {
        const int col  = n0g + j * 16 + lm;
        const float bias = down_b[e * DOWN + col];
        #pragma unroll
        for (int r = 0; r < 8; ++r) {
            if (slot[r] >= 0)
                hbuf[(size_t)slot[r] * DOWN + col] = prob[r] * (acc[j][r] + bias);
        }
    }
}

// ---------------- combine slots + gelu(tanh) -> bf16 ----------------
__device__ __forceinline__ float gelu_new(float v) {
    const float c = 0.7978845608028654f;
    return 0.5f * v * (1.f + tanhf(c * (v + 0.044715f * v * v * v)));
}

__global__ __launch_bounds__(256) void combine_gelu_kernel(const float* __restrict__ hbuf,
                                                           __bf16* __restrict__ hbf) {
    const size_t base = ((size_t)blockIdx.x * 256 + threadIdx.x) * 4;
    const size_t n = base >> 9;          // /512
    const size_t d = base & 511;
    const float4 a = *(const float4*)(hbuf + n * (2 * DOWN) + d);
    const float4 b = *(const float4*)(hbuf + n * (2 * DOWN) + DOWN + d);
    v4bf o;
    o[0] = (__bf16)gelu_new(a.x + b.x);
    o[1] = (__bf16)gelu_new(a.y + b.y);
    o[2] = (__bf16)gelu_new(a.z + b.z);
    o[3] = (__bf16)gelu_new(a.w + b.w);
    *(v4bf*)(hbf + n * DOWN + d) = o;
}

// ---------------- up-projection GEMM (double-buffered async-LDS B) ----------
// grid (HIDDEN/64, NTOK/64), block 128 (4 waves). Block covers 64 output cols
// x 64 rows; wave w handles rows blockIdx.y*64 + w*16. K = 512.
__global__ __launch_bounds__(128) void up_gemm_kernel(const __bf16* __restrict__ hbf,
                                                      const __bf16* __restrict__ uwbf,
                                                      const float* __restrict__ up_b,
                                                      float* __restrict__ out) {
    __shared__ __align__(16) char sB[2][64 * PADB];
    const int wv   = threadIdx.x >> 5;
    const int lane = threadIdx.x & 31;
    const int lm   = lane & 15;
    const int lh   = lane >> 4;
    const int col0 = blockIdx.x * 64;
    const int row0 = blockIdx.y * 64 + wv * 16;
    const __bf16* arow  = hbf + (size_t)(row0 + lm) * DOWN + lh * 16;
    const __bf16* ubase = uwbf + (size_t)col0 * DOWN;
    const unsigned int ldsBase0 = (unsigned int)(uintptr_t)(void*)sB[0];
    const unsigned int ldsBase1 = (unsigned int)(uintptr_t)(void*)sB[1];

    v8f acc[4] = {};
    stage_b_chunk<DOWN>(ubase, 0, ldsBase0, threadIdx.x);
    for (int c = 0, k0 = 0; k0 < DOWN; k0 += 64, ++c) {
        wait_async0();
        __syncthreads();
        if (k0 + 64 < DOWN)
            stage_b_chunk<DOWN>(ubase, k0 + 64,
                                (c & 1) ? ldsBase0 : ldsBase1, threadIdx.x);
        const char* bb = sB[c & 1];
        if (k0 + 64 < DOWN) __builtin_prefetch(arow + k0 + 64, 0, 3);
        #pragma unroll
        for (int kk = 0; kk < 64; kk += 32) {
            const v16bf a = make_afrag(arow + k0 + kk);
            #pragma unroll
            for (int j = 0; j < 4; ++j) {
                const v16bf b = *(const v16bf*)(bb + (j * 16 + lm) * PADB + kk * 2 + lh * 32);
                acc[j] = wmma_bf16(a, b, acc[j]);
            }
        }
    }

    #pragma unroll
    for (int j = 0; j < 4; ++j) {
        const int col  = col0 + j * 16 + lm;
        const float bias = up_b[col];
        #pragma unroll
        for (int r = 0; r < 8; ++r) {
            const int row = row0 + r + 8 * lh;
            out[(size_t)row * HIDDEN + col] = acc[j][r] + bias;
        }
    }
}

// ---------------- load-balance loss (deterministic reductions) ------------
__global__ __launch_bounds__(256) void loss_sums_kernel(const int* __restrict__ ti,
                                                        const float* __restrict__ tp,
                                                        float* __restrict__ Sbuf) {
    __shared__ float red[256];
    const int e = blockIdx.x;
    float s = 0.f;
    for (int j = threadIdx.x; j < NTOK * TOPK; j += 256)
        if (ti[j] == e) s += tp[j];
    red[threadIdx.x] = s;
    __syncthreads();
    for (int off = 128; off > 0; off >>= 1) {
        if (threadIdx.x < off) red[threadIdx.x] += red[threadIdx.x + off];
        __syncthreads();
    }
    if (threadIdx.x == 0) Sbuf[e] = red[0];
}

__global__ void loss_final_kernel(const float* __restrict__ Sbuf, float* __restrict__ loss_out) {
    const int lane = threadIdx.x & 31;
    float v = (lane < NEXP) ? Sbuf[lane] * Sbuf[lane] : 0.f;
    #pragma unroll
    for (int off = 16; off > 0; off >>= 1)
        v += __shfl_xor(v, off, 32);
    if (lane == 0) {
        const float invN2 = 1.f / ((float)NTOK * (float)NTOK);
        loss_out[0] = (float)NEXP * 0.1f * v * invN2;
    }
}

// ---------------- host-side orchestration ----------------
static inline size_t align256(size_t x) { return (x + 255) & ~(size_t)255; }

extern "C" void kernel_launch(void* const* d_in, const int* in_sizes, int n_in,
                              void* d_out, int out_size, void* d_ws, size_t ws_size,
                              hipStream_t stream) {
    (void)in_sizes; (void)n_in; (void)ws_size; (void)out_size;
    const float* x      = (const float*)d_in[0];   // [N, 2048]
    const float* gate_w = (const float*)d_in[1];   // [16, 2048]
    const float* down_w = (const float*)d_in[2];   // [16, 512, 2048]
    const float* down_b = (const float*)d_in[3];   // [16, 512]
    const float* up_w   = (const float*)d_in[4];   // [2048, 512]
    const float* up_b   = (const float*)d_in[5];   // [2048]
    float* out = (float*)d_out;                    // [N*2048] floats + 1 loss

    // workspace carve-up
    char* ws = (char*)d_ws;
    size_t off = 0;
    int*    counts     = (int*)(ws + off);  off = align256(off + NEXP * sizeof(int));
    float*  Sbuf       = (float*)(ws + off); off = align256(off + NEXP * sizeof(float));
    int*    tokenIds   = (int*)(ws + off);  off = align256(off + (size_t)NEXP * NTOK * sizeof(int));
    int*    slotIds    = (int*)(ws + off);  off = align256(off + (size_t)NEXP * NTOK * sizeof(int));
    float*  tokenProbs = (float*)(ws + off); off = align256(off + (size_t)NEXP * NTOK * sizeof(float));
    int*    ti         = (int*)(ws + off);  off = align256(off + (size_t)NTOK * TOPK * sizeof(int));
    float*  tp         = (float*)(ws + off); off = align256(off + (size_t)NTOK * TOPK * sizeof(float));
    __bf16* xbf        = (__bf16*)(ws + off); off = align256(off + (size_t)NTOK * HIDDEN * 2);
    __bf16* dwbf       = (__bf16*)(ws + off); off = align256(off + (size_t)NEXP * DOWN * HIDDEN * 2);
    __bf16* uwbf       = (__bf16*)(ws + off); off = align256(off + (size_t)HIDDEN * DOWN * 2);
    float*  hbuf       = (float*)(ws + off); off = align256(off + (size_t)NTOK * TOPK * DOWN * sizeof(float));
    __bf16* hbf        = (__bf16*)(ws + off); off = align256(off + (size_t)NTOK * DOWN * 2);

    // 1) zero expert counters
    init_kernel<<<1, 32, 0, stream>>>(counts);

    // 2) f32 -> bf16 preconversion (x, down_w, up_w)
    {
        int n4 = NTOK * HIDDEN / 4;
        cvt_bf16_kernel<<<(n4 + 255) / 256, 256, 0, stream>>>(x, xbf, n4);
        n4 = NEXP * DOWN * HIDDEN / 4;
        cvt_bf16_kernel<<<(n4 + 255) / 256, 256, 0, stream>>>(down_w, dwbf, n4);
        n4 = HIDDEN * DOWN / 4;
        cvt_bf16_kernel<<<(n4 + 255) / 256, 256, 0, stream>>>(up_w, uwbf, n4);
    }

    // 3) gating + expert binning
    gate_kernel<<<NTOK, 512, 0, stream>>>(x, gate_w, counts, tokenIds, slotIds,
                                          tokenProbs, ti, tp);

    // 4) gathered per-expert down-projection (WMMA bf16, double-buffered async LDS)
    {
        dim3 grid(NEXP, DOWN / 64, 128);
        down_gemm_kernel<<<grid, 128, 0, stream>>>(xbf, dwbf, down_b, counts,
                                                   tokenIds, slotIds, tokenProbs, hbuf);
    }

    // 5) combine two expert slots + gelu -> bf16
    combine_gelu_kernel<<<(NTOK * DOWN / 4) / 256, 256, 0, stream>>>(hbuf, hbf);

    // 6) up-projection GEMM (WMMA bf16, double-buffered async LDS)
    {
        dim3 grid(HIDDEN / 64, NTOK / 64);
        up_gemm_kernel<<<grid, 128, 0, stream>>>(hbf, uwbf, up_b, out);
    }

    // 7) load-balancing loss
    loss_sums_kernel<<<NEXP, 256, 0, stream>>>(ti, tp, Sbuf);
    loss_final_kernel<<<1, 32, 0, stream>>>(Sbuf, out + (size_t)NTOK * HIDDEN);
}
